// BiLSTM_CRF_35012573397370
// MI455X (gfx1250) — matrix operand order repaired and checked by reading, not compiled, not run on hardware
//
#include <hip/hip_runtime.h>
#include <hip/hip_bf16.h>
#include <stdint.h>

#define EMBD   256
#define HID    512
#define KTAG   32
#define TSEQ   256
#define BATCH  128
#define START_TAG 30
#define STOP_TAG  31
#define NEGINF (-10000.0f)
#define G4 (4 * HID)

// Padded LDS weight-tile layout (pad = 4 dwords per row via TDM pad feature):
// keeps every fragment base 16B-aligned (ds_load_b128) and rotates banks by 4
// per row so 16 lanes reading 16 different rows stay conflict-free.
// W_hh rows: 512 f16 = 1024B + 16B pad ; W_ih rows: 256 f16 = 512B + 16B pad.
#define WHH_ROW_BYTES 1040
#define WIH_ROW_BYTES 528
#define WHH_BYTES (64 * WHH_ROW_BYTES)               // 66560
#define WIH_OFF   WHH_BYTES
#define SMEM_BYTES (WHH_BYTES + 64 * WIH_ROW_BYTES)  // 100352

typedef __attribute__((ext_vector_type(16))) _Float16 v16h;
typedef __attribute__((ext_vector_type(8)))  float    v8f;
typedef __attribute__((ext_vector_type(4)))  unsigned int u32x4;
typedef __attribute__((ext_vector_type(8)))  int      i32x8;
typedef __attribute__((ext_vector_type(4)))  int      i32x4;

#if __has_builtin(__builtin_amdgcn_tensor_load_to_lds)
#define HAVE_TDM 1
#else
#define HAVE_TDM 0
#endif

union FragH { uint4 q[2]; v16h h; };

// A-matrix fragment (16x32 f16): lane holds row M = lane&15; K pairs:
// v[0..7] = k0 + (lane>>4)*16 + {0..15} (contiguous).
// Both 16-byte groups are 16B-aligned -> global_load_b128.
__device__ __forceinline__ v16h load_frag_A(const _Float16* __restrict__ S,
                                            int ldk, int rowbase, int k0, int lane) {
  const int m = lane & 15, kh = lane >> 4;
  const _Float16* base = S + (size_t)(rowbase + m) * ldk + k0 + kh * 16;
  FragH f;
  f.q[0] = *reinterpret_cast<const uint4*>(base);
  f.q[1] = *reinterpret_cast<const uint4*>(base + 8);
  return f.h;
}

// B-matrix fragment (32x16 f16) for C = A * W^T, from global (row of W
// = output column). 32 contiguous bytes, 16B-aligned -> 2x global_load_b128.
__device__ __forceinline__ v16h load_frag_B(const _Float16* __restrict__ W,
                                            int ldk, int colbase, int k0, int lane) {
  const int n = lane & 15, kh = lane >> 4;
  const _Float16* base = W + (size_t)(colbase + n) * ldk + k0 + kh * 16;
  FragH f;
  f.q[0] = *reinterpret_cast<const uint4*>(base);
  f.q[1] = *reinterpret_cast<const uint4*>(base + 8);
  return f.h;
}

// B-matrix fragment from a padded LDS tile (row_bytes stride, 16 rows).
// 16B-aligned -> 2x ds_load_b128.
__device__ __forceinline__ v16h load_frag_B_lds(const char* tile, int row_bytes,
                                                int k0, int lane) {
  const int n = lane & 15, kh = lane >> 4;
  const char* base = tile + (size_t)n * row_bytes + (size_t)(k0 + kh * 16) * 2;
  FragH f;
  f.q[0] = *reinterpret_cast<const uint4*>(base);
  f.q[1] = *reinterpret_cast<const uint4*>(base + 16);
  return f.h;
}

__device__ __forceinline__ v8f wmma_f16(v16h a, v16h b, v8f c) {
  return __builtin_amdgcn_wmma_f32_16x16x32_f16(false, a, false, b,
                                                (short)0, c, false, false);
}

__device__ __forceinline__ float sigmoidf_(float x) {
  return 1.0f / (1.0f + __expf(-x));
}

// ---------------------------------------------------------------------------
// Tensor Data Mover: 2D f16 tile (width x rows, rows of stride_elems in
// global) -> LDS with 4-dword pad after each row (pad_amount=3).
// pad_interval: 7 == every 256 dwords (1024B row), 6 == every 128 dwords.
// D# per CDNA5 ISA. 6-arg builtin (clang-23).
// ---------------------------------------------------------------------------
__device__ __forceinline__ void tdm_load_2d(void* lds_dst, const void* gsrc,
                                            uint32_t width_elems, uint32_t rows,
                                            uint32_t stride_elems,
                                            uint32_t pad_interval) {
#if HAVE_TDM
  const uint64_t ga = (uint64_t)(uintptr_t)gsrc;
  u32x4 g0;
  g0[0] = 0x1u;                                        // count=1, user D#
  g0[1] = (uint32_t)(uintptr_t)lds_dst;                // lds_addr (bytes)
  g0[2] = (uint32_t)ga;                                // global_addr[31:0]
  g0[3] = (uint32_t)((ga >> 32) & 0x1FFFFFFu) | (2u << 30);  // [56:32] | type=2
  i32x8 g1;
  g1[0] = (int)((1u << 16) | (1u << 20) | (pad_interval << 22) | (3u << 25));
  //            data_size=2B  pad_enable   pad_interval          pad=4 dwords
  g1[1] = (int)((width_elems & 0xFFFFu) << 16);                  // tensor_dim0 lo
  g1[2] = (int)(((width_elems >> 16) & 0xFFFFu) | ((rows & 0xFFFFu) << 16)); // dim0 hi | dim1 lo
  g1[3] = (int)(((rows >> 16) & 0xFFFFu) | ((width_elems & 0xFFFFu) << 16)); // dim1 hi | tile_dim0
  g1[4] = (int)(rows & 0xFFFFu);                                 // tile_dim1 | tile_dim2=0
  g1[5] = (int)stride_elems;                                     // tensor_dim0_stride lo
  g1[6] = 0;                                                     // stride0 hi | stride1 lo
  g1[7] = 0;
  const i32x4 z4 = {0, 0, 0, 0};
  const i32x8 z8 = {0, 0, 0, 0, 0, 0, 0, 0};
  __builtin_amdgcn_tensor_load_to_lds(g0, g1, z4, z4, z8, 0);
#else
  (void)lds_dst; (void)gsrc; (void)width_elems; (void)rows;
  (void)stride_elems; (void)pad_interval;
#endif
}

// ---------------------------------------------------------------------------
// Embedding gather + f32 -> f16
// ---------------------------------------------------------------------------
__global__ __launch_bounds__(256) void gather_embed_kernel(
    const int* __restrict__ sent, const float* __restrict__ emb,
    _Float16* __restrict__ out) {
  const int tb = blockIdx.x * 4 + (threadIdx.x >> 6);
  const int e  = (threadIdx.x & 63) * 4;
  const int idx = sent[tb];
  const float4 v = *reinterpret_cast<const float4*>(emb + (size_t)idx * EMBD + e);
  union { _Float16 h[4]; uint64_t u; } pk;
  pk.h[0] = (_Float16)v.x; pk.h[1] = (_Float16)v.y;
  pk.h[2] = (_Float16)v.z; pk.h[3] = (_Float16)v.w;
  *reinterpret_cast<uint64_t*>(out + (size_t)tb * EMBD + e) = pk.u;
}

__global__ __launch_bounds__(256) void f32_to_f16_kernel(
    const float* __restrict__ in, _Float16* __restrict__ out, int n) {
  const int i = blockIdx.x * 256 + threadIdx.x;
  if (i < n) out[i] = (_Float16)in[i];
}

// ---------------------------------------------------------------------------
// One BiLSTM time step, both directions (gridDim.y == 2).
// Weight tiles for this block's 16 hidden units (x4 gates) are staged into
// LDS once per block via the Tensor Data Mover, then every wave reads its B
// fragments from LDS (ds_load_b128) while streaming per-wave A rows from
// global (global_load_b128).
// ---------------------------------------------------------------------------
__global__ __launch_bounds__(256) void lstm_step_kernel(
    const _Float16* __restrict__ embeds,
    const _Float16* __restrict__ wih_f, const _Float16* __restrict__ whh_f,
    const _Float16* __restrict__ wih_b, const _Float16* __restrict__ whh_b,
    const float* __restrict__ bih_f, const float* __restrict__ bhh_f,
    const float* __restrict__ bih_b, const float* __restrict__ bhh_b,
    _Float16* __restrict__ hs_f, _Float16* __restrict__ hs_b,
    float* __restrict__ cstate, int t) {
  extern __shared__ char smem[];
  const int dir   = blockIdx.y;
  const int t_eff = dir ? (TSEQ - 1 - t) : t;
  const _Float16* Wih = dir ? wih_b : wih_f;
  const _Float16* Whh = dir ? whh_b : whh_f;
  const float* bih = dir ? bih_b : bih_f;
  const float* bhh = dir ? bhh_b : bhh_f;
  _Float16* hs = dir ? hs_b : hs_f;
  float* cst = cstate + (size_t)dir * BATCH * HID;
  const int prev_t = dir ? (t_eff + 1) : (t_eff - 1);   // only read when t > 0
  const _Float16* hprev = hs + (size_t)prev_t * BATCH * HID;
  const _Float16* xrow  = embeds + (size_t)t_eff * BATCH * EMBD;

  const int wave = threadIdx.x >> 5;
  const int lane = threadIdx.x & 31;
  const int rowbase = wave * 16;        // batch rows [rowbase, rowbase+16)
  const int hg = blockIdx.x;            // hidden-unit group of 16
  const int n  = lane & 15;
  const int kh = lane >> 4;
  const int hc = hg * 16 + n;

#if HAVE_TDM
  if (wave == 0) {                      // TDM is EXEC-independent; issue once
#pragma unroll
    for (int g = 0; g < 4; ++g) {
      tdm_load_2d(smem + (size_t)g * 16 * WHH_ROW_BYTES,
                  Whh + (size_t)(g * HID + hg * 16) * HID, HID, 16, HID, 7);
      tdm_load_2d(smem + WIH_OFF + (size_t)g * 16 * WIH_ROW_BYTES,
                  Wih + (size_t)(g * HID + hg * 16) * EMBD, EMBD, 16, EMBD, 6);
    }
    __builtin_amdgcn_s_wait_tensorcnt(0);
  }
#else
  // Cooperative fallback copy into the same padded layout.
  for (int idx = threadIdx.x; idx < 64 * 256; idx += 256) {   // W_hh dwords
    const int r = idx >> 8, d = idx & 255;
    const int grow = (r >> 4) * HID + hg * 16 + (r & 15);
    *reinterpret_cast<uint32_t*>(smem + (size_t)r * WHH_ROW_BYTES + d * 4) =
        *reinterpret_cast<const uint32_t*>(
            reinterpret_cast<const char*>(Whh) + (size_t)grow * HID * 2 + d * 4);
  }
  for (int idx = threadIdx.x; idx < 64 * 128; idx += 256) {   // W_ih dwords
    const int r = idx >> 7, d = idx & 127;
    const int grow = (r >> 4) * HID + hg * 16 + (r & 15);
    *reinterpret_cast<uint32_t*>(smem + WIH_OFF + (size_t)r * WIH_ROW_BYTES + d * 4) =
        *reinterpret_cast<const uint32_t*>(
            reinterpret_cast<const char*>(Wih) + (size_t)grow * EMBD * 2 + d * 4);
  }
#endif
  __syncthreads();

  const char* whh_s = smem;
  const char* wih_s = smem + WIH_OFF;

  v8f acc0 = {}, acc1 = {}, acc2 = {}, acc3 = {};   // i, f, g, o

  if (t > 0) {                           // recurrent part: K = HID
    for (int k0 = 0; k0 < HID; k0 += 32) {
      v16h a  = load_frag_A(hprev, HID, rowbase, k0, lane);
      v16h b0 = load_frag_B_lds(whh_s + 0 * 16 * WHH_ROW_BYTES, WHH_ROW_BYTES, k0, lane);
      v16h b1 = load_frag_B_lds(whh_s + 1 * 16 * WHH_ROW_BYTES, WHH_ROW_BYTES, k0, lane);
      v16h b2 = load_frag_B_lds(whh_s + 2 * 16 * WHH_ROW_BYTES, WHH_ROW_BYTES, k0, lane);
      v16h b3 = load_frag_B_lds(whh_s + 3 * 16 * WHH_ROW_BYTES, WHH_ROW_BYTES, k0, lane);
      acc0 = wmma_f16(a, b0, acc0);
      acc1 = wmma_f16(a, b1, acc1);
      acc2 = wmma_f16(a, b2, acc2);
      acc3 = wmma_f16(a, b3, acc3);
    }
  }
  for (int k0 = 0; k0 < EMBD; k0 += 32) {  // input part: K = EMBD
    if (k0 + 32 < EMBD)
      __builtin_prefetch(xrow + (size_t)(rowbase + n) * EMBD + k0 + 32, 0, 1);
    v16h a  = load_frag_A(xrow, EMBD, rowbase, k0, lane);
    v16h b0 = load_frag_B_lds(wih_s + 0 * 16 * WIH_ROW_BYTES, WIH_ROW_BYTES, k0, lane);
    v16h b1 = load_frag_B_lds(wih_s + 1 * 16 * WIH_ROW_BYTES, WIH_ROW_BYTES, k0, lane);
    v16h b2 = load_frag_B_lds(wih_s + 2 * 16 * WIH_ROW_BYTES, WIH_ROW_BYTES, k0, lane);
    v16h b3 = load_frag_B_lds(wih_s + 3 * 16 * WIH_ROW_BYTES, WIH_ROW_BYTES, k0, lane);
    acc0 = wmma_f16(a, b0, acc0);
    acc1 = wmma_f16(a, b1, acc1);
    acc2 = wmma_f16(a, b2, acc2);
    acc3 = wmma_f16(a, b3, acc3);
  }

  const float bias0 = bih[0 * HID + hc] + bhh[0 * HID + hc];
  const float bias1 = bih[1 * HID + hc] + bhh[1 * HID + hc];
  const float bias2 = bih[2 * HID + hc] + bhh[2 * HID + hc];
  const float bias3 = bih[3 * HID + hc] + bhh[3 * HID + hc];

#pragma unroll
  for (int v = 0; v < 8; ++v) {
    const int b = rowbase + v + 8 * kh;   // D layout: row = v + 8*(lane>=16)
    const float gi = acc0[v] + bias0;
    const float gf = acc1[v] + bias1;
    const float gg = acc2[v] + bias2;
    const float go = acc3[v] + bias3;
    const float cp = (t == 0) ? 0.0f : cst[(size_t)b * HID + hc];
    const float cn = sigmoidf_(gf) * cp + sigmoidf_(gi) * tanhf(gg);
    const float hn = sigmoidf_(go) * tanhf(cn);
    cst[(size_t)b * HID + hc] = cn;
    hs[((size_t)t_eff * BATCH + b) * HID + hc] = (_Float16)hn;
  }
}

// ---------------------------------------------------------------------------
// feats = [h_f | h_b] * W_out^T + b_out   (M = T*B, N = 32, K = 1024)
// ---------------------------------------------------------------------------
__global__ __launch_bounds__(256) void feats_kernel(
    const _Float16* __restrict__ hs_f, const _Float16* __restrict__ hs_b,
    const _Float16* __restrict__ wout, const float* __restrict__ bout,
    float* __restrict__ feats) {
  const int wave = threadIdx.x >> 5, lane = threadIdx.x & 31;
  const int mbase   = blockIdx.x * 64 + (wave >> 1) * 16;
  const int colbase = (wave & 1) * 16;
  v8f acc = {};
  for (int k0 = 0; k0 < HID; k0 += 32) {
    v16h a = load_frag_A(hs_f, HID, mbase, k0, lane);
    v16h b = load_frag_B(wout, 2 * HID, colbase, k0, lane);
    acc = wmma_f16(a, b, acc);
  }
  for (int k0 = 0; k0 < HID; k0 += 32) {
    v16h a = load_frag_A(hs_b, HID, mbase, k0, lane);
    v16h b = load_frag_B(wout, 2 * HID, colbase, HID + k0, lane);
    acc = wmma_f16(a, b, acc);
  }
  const int nn = lane & 15, kh = lane >> 4;
  const float bb = bout[colbase + nn];
#pragma unroll
  for (int v = 0; v < 8; ++v) {
    const int m = mbase + v + 8 * kh;
    feats[(size_t)m * KTAG + colbase + nn] = acc[v] + bb;
  }
}

// ---------------------------------------------------------------------------
// Viterbi: single 1024-thread block, fv + transitions in LDS, uint8 backptrs.
// Output: out[0..B) = score, out[B + b*T + t] = path tag (as float).
// ---------------------------------------------------------------------------
__global__ __launch_bounds__(1024) void viterbi_kernel(
    const float* __restrict__ feats, const float* __restrict__ trans,
    uint8_t* __restrict__ bp, float* __restrict__ out) {
  __shared__ float fv[BATCH][KTAG];   // 16 KB
  __shared__ float tr[KTAG][KTAG];    //  4 KB
  const int tid = threadIdx.x;
  tr[tid >> 5][tid & 31] = trans[tid];          // K*K == 1024 == blockDim
  for (int p = tid; p < BATCH * KTAG; p += 1024)
    fv[p >> 5][p & 31] = ((p & 31) == START_TAG) ? 0.0f : NEGINF;
  __syncthreads();

  for (int t = 0; t < TSEQ; ++t) {
    float nv[4];
#pragma unroll
    for (int q = 0; q < 4; ++q) {
      const int p = tid + q * 1024, b = p >> 5, j = p & 31;
      float best = -3.4e38f; int arg = 0;
      for (int i = 0; i < KTAG; ++i) {
        const float v = fv[b][i] + tr[j][i];
        if (v > best) { best = v; arg = i; }    // strict > keeps first argmax
      }
      nv[q] = best + feats[((size_t)t * BATCH + b) * KTAG + j];
      bp[((size_t)t * BATCH + b) * KTAG + j] = (uint8_t)arg;
    }
    __syncthreads();
#pragma unroll
    for (int q = 0; q < 4; ++q) {
      const int p = tid + q * 1024;
      fv[p >> 5][p & 31] = nv[q];
    }
    __syncthreads();
  }

  if (tid < BATCH) {
    const int b = tid;
    float best = -3.4e38f; int arg = 0;
    for (int j = 0; j < KTAG; ++j) {
      const float v = fv[b][j] + tr[STOP_TAG][j];
      if (v > best) { best = v; arg = j; }
    }
    out[b] = best;
    int tag = arg;
    for (int t = TSEQ - 1; t >= 0; --t) {
      out[BATCH + (size_t)b * TSEQ + t] = (float)tag;
      tag = bp[((size_t)t * BATCH + b) * KTAG + tag];
    }
  }
}

// ---------------------------------------------------------------------------
extern "C" void kernel_launch(void* const* d_in, const int* in_sizes, int n_in,
                              void* d_out, int out_size, void* d_ws, size_t ws_size,
                              hipStream_t stream) {
  (void)in_sizes; (void)n_in; (void)out_size; (void)ws_size;

  const int*   sent   = (const int*)  d_in[0];
  const float* emb    = (const float*)d_in[1];
  const float* W_ih_f = (const float*)d_in[2];
  const float* W_hh_f = (const float*)d_in[3];
  const float* b_ih_f = (const float*)d_in[4];
  const float* b_hh_f = (const float*)d_in[5];
  const float* W_ih_b = (const float*)d_in[6];
  const float* W_hh_b = (const float*)d_in[7];
  const float* b_ih_b = (const float*)d_in[8];
  const float* b_hh_b = (const float*)d_in[9];
  const float* W_out  = (const float*)d_in[10];
  const float* b_out  = (const float*)d_in[11];
  const float* trans  = (const float*)d_in[12];

  char* ws = (char*)d_ws;
  size_t off = 0;
  auto alloc = [&](size_t bytes) -> char* {
    char* p = ws + off;
    off += (bytes + 255) & ~(size_t)255;
    return p;
  };
  _Float16* embeds  = (_Float16*)alloc((size_t)TSEQ * BATCH * EMBD * 2);
  _Float16* wih16_f = (_Float16*)alloc((size_t)G4 * EMBD * 2);
  _Float16* whh16_f = (_Float16*)alloc((size_t)G4 * HID * 2);
  _Float16* wih16_b = (_Float16*)alloc((size_t)G4 * EMBD * 2);
  _Float16* whh16_b = (_Float16*)alloc((size_t)G4 * HID * 2);
  _Float16* wout16  = (_Float16*)alloc((size_t)KTAG * 2 * HID * 2);
  _Float16* hs_f    = (_Float16*)alloc((size_t)TSEQ * BATCH * HID * 2);
  _Float16* hs_b    = (_Float16*)alloc((size_t)TSEQ * BATCH * HID * 2);
  float*    cstate  = (float*)   alloc((size_t)2 * BATCH * HID * 4);
  float*    feats   = (float*)   alloc((size_t)TSEQ * BATCH * KTAG * 4);
  uint8_t*  bp      = (uint8_t*) alloc((size_t)TSEQ * BATCH * KTAG);

  gather_embed_kernel<<<TSEQ * BATCH / 4, 256, 0, stream>>>(sent, emb, embeds);
  {
    int n;
    n = G4 * EMBD; f32_to_f16_kernel<<<(n + 255) / 256, 256, 0, stream>>>(W_ih_f, wih16_f, n);
    n = G4 * HID;  f32_to_f16_kernel<<<(n + 255) / 256, 256, 0, stream>>>(W_hh_f, whh16_f, n);
    n = G4 * EMBD; f32_to_f16_kernel<<<(n + 255) / 256, 256, 0, stream>>>(W_ih_b, wih16_b, n);
    n = G4 * HID;  f32_to_f16_kernel<<<(n + 255) / 256, 256, 0, stream>>>(W_hh_b, whh16_b, n);
    n = KTAG * 2 * HID; f32_to_f16_kernel<<<(n + 255) / 256, 256, 0, stream>>>(W_out, wout16, n);
  }

  for (int t = 0; t < TSEQ; ++t) {
    lstm_step_kernel<<<dim3(HID / 16, 2), 256, SMEM_BYTES, stream>>>(
        embeds, wih16_f, whh16_f, wih16_b, whh16_b,
        b_ih_f, b_hh_f, b_ih_b, b_hh_b, hs_f, hs_b, cstate, t);
  }

  feats_kernel<<<TSEQ * BATCH / 64, 256, 0, stream>>>(hs_f, hs_b, wout16, b_out, feats);
  viterbi_kernel<<<1, 1024, 0, stream>>>(feats, trans, bp, (float*)d_out);
}
